// SelfAttentionModuleV2_74706661147106
// MI455X (gfx1250) — matrix operand order repaired and checked by reading, not compiled, not run on hardware
//
#include <hip/hip_runtime.h>
#include <hip/hip_bf16.h>

typedef __attribute__((ext_vector_type(16))) __bf16 v16bf;
typedef __attribute__((ext_vector_type(8)))  float  v8f;

#define BN_EPS 1e-5f

__device__ __forceinline__ unsigned short f2bf(float f) {
  unsigned int u = __float_as_uint(f);
  unsigned int r = (u + 0x7FFFu + ((u >> 16) & 1u)) >> 16;  // round-to-nearest-even
  return (unsigned short)r;
}
__device__ __forceinline__ float bf2f(unsigned short s) {
  return __uint_as_float(((unsigned int)s) << 16);
}

union FragAB { v16bf v; uint4 q[2]; };
union Pack8  { uint4 q; unsigned short u[8]; };

// ---------------------------------------------------------------------------
// small prep kernels
// ---------------------------------------------------------------------------
__global__ void __launch_bounds__(256)
cast_bf16_kernel(const float* __restrict__ s, unsigned short* __restrict__ d, int n) {
  int i = blockIdx.x * 256 + threadIdx.x;
  if (i < n) d[i] = f2bf(s[i]);
}

// x: [B][C][HW] f32  ->  xT: [B][HW][C] bf16, LDS-tiled 32x32 transpose.
// Reads: 32 consecutive f32 per row (128B coalesced). Writes: 32 consecutive
// bf16 per row (64B coalesced). +1 padding kills LDS bank conflicts.
__global__ void __launch_bounds__(256)
transpose_cast_kernel(const float* __restrict__ x, unsigned short* __restrict__ xT,
                      int C, int HW) {
  __shared__ unsigned short tile[32][33];
  const int n0 = blockIdx.x * 32;
  const int c0 = blockIdx.y * 32;
  const int b  = blockIdx.z;
  const int ti = threadIdx.x & 31;   // inner (contiguous) index
  const int tr = threadIdx.x >> 5;   // 0..7
#pragma unroll
  for (int rr = 0; rr < 4; ++rr) {
    const int c = tr + rr * 8;
    tile[c][ti] = f2bf(x[((long long)b * C + c0 + c) * HW + n0 + ti]);
  }
  __syncthreads();
#pragma unroll
  for (int rr = 0; rr < 4; ++rr) {
    const int n = tr + rr * 8;
    xT[((long long)b * HW + n0 + n) * C + c0 + ti] = tile[ti][n];
  }
}

// ---------------------------------------------------------------------------
// bf16 WMMA GEMM:  D[m][n] = sum_k A[m][k] * Bt[z][n][k]   (+ bias / BN+ReLU)
//   A  : [M][K] bf16 (weights, row-major)
//   Bt : [Z][N][K] bf16 (activations, pixel-major => B-fragments contiguous)
// mode 0: out f32  [Z][N][M] (transposed, bias)
// mode 1: out bf16 [Z][N][M] (transposed, bias + BN + ReLU when bng != null)
// mode 2: out f32  [Z][M][N] (standard NCHW, bias)  -> final result layout
// Block: 256 thr = 8 waves; block tile M=64, N=256; wave tile 32x64 (8 wmma/K-step)
// ---------------------------------------------------------------------------
__global__ void __launch_bounds__(256)
gemm_wmma_bf16(const unsigned short* __restrict__ A,
               const unsigned short* __restrict__ Bt,
               float* __restrict__ outF,
               unsigned short* __restrict__ outB,
               const float* __restrict__ bias,
               const float* __restrict__ bng, const float* __restrict__ bnb,
               const float* __restrict__ bnm, const float* __restrict__ bnv,
               int M, int K, int N, int mode) {
  const int tid  = threadIdx.x;
  const int wave = tid >> 5, lane = tid & 31;
  const int ln = lane & 15, kh = lane >> 4;
  const int mw = wave & 1, nw = wave >> 1;
  const int mBase = blockIdx.y * 64 + mw * 32;
  const int nBase = blockIdx.x * 256 + nw * 64;
  const long long zB = (long long)blockIdx.z * N * K;

  const unsigned short* a0p = A + (long long)(mBase + ln) * K;
  const unsigned short* a1p = A + (long long)(mBase + 16 + ln) * K;
  const unsigned short* b0p = Bt + zB + (long long)(nBase +  0 + ln) * K;
  const unsigned short* b1p = Bt + zB + (long long)(nBase + 16 + ln) * K;
  const unsigned short* b2p = Bt + zB + (long long)(nBase + 32 + ln) * K;
  const unsigned short* b3p = Bt + zB + (long long)(nBase + 48 + ln) * K;

  v8f acc[2][4];
#pragma unroll
  for (int i = 0; i < 2; ++i)
#pragma unroll
    for (int j = 0; j < 4; ++j)
      acc[i][j] = (v8f){0.f, 0.f, 0.f, 0.f, 0.f, 0.f, 0.f, 0.f};

  // A-frag (16x32 bf16): elems 0..7 at K = kh*8 + e ; elems 8..15 at K = 16 + kh*8 + e
  const int aoff0 = kh * 8, aoff1 = 16 + kh * 8;
  // B-frag (32x16 bf16): elems 0..7 at K = kh*8 + e ; elems 8..15 at K = 16 + kh*8 + e
  const int boff0 = kh * 8, boff1 = 16 + kh * 8;

  for (int k0 = 0; k0 < K; k0 += 32) {
    FragAB a0, a1, b0, b1, b2, b3;
    a0.q[0] = *(const uint4*)(a0p + k0 + aoff0);
    a0.q[1] = *(const uint4*)(a0p + k0 + aoff1);
    a1.q[0] = *(const uint4*)(a1p + k0 + aoff0);
    a1.q[1] = *(const uint4*)(a1p + k0 + aoff1);
    b0.q[0] = *(const uint4*)(b0p + k0 + boff0);
    b0.q[1] = *(const uint4*)(b0p + k0 + boff1);
    b1.q[0] = *(const uint4*)(b1p + k0 + boff0);
    b1.q[1] = *(const uint4*)(b1p + k0 + boff1);
    b2.q[0] = *(const uint4*)(b2p + k0 + boff0);
    b2.q[1] = *(const uint4*)(b2p + k0 + boff1);
    b3.q[0] = *(const uint4*)(b3p + k0 + boff0);
    b3.q[1] = *(const uint4*)(b3p + k0 + boff1);

    acc[0][0] = __builtin_amdgcn_wmma_f32_16x16x32_bf16(false, a0.v, false, b0.v, (short)0, acc[0][0], false, false);
    acc[0][1] = __builtin_amdgcn_wmma_f32_16x16x32_bf16(false, a0.v, false, b1.v, (short)0, acc[0][1], false, false);
    acc[0][2] = __builtin_amdgcn_wmma_f32_16x16x32_bf16(false, a0.v, false, b2.v, (short)0, acc[0][2], false, false);
    acc[0][3] = __builtin_amdgcn_wmma_f32_16x16x32_bf16(false, a0.v, false, b3.v, (short)0, acc[0][3], false, false);
    acc[1][0] = __builtin_amdgcn_wmma_f32_16x16x32_bf16(false, a1.v, false, b0.v, (short)0, acc[1][0], false, false);
    acc[1][1] = __builtin_amdgcn_wmma_f32_16x16x32_bf16(false, a1.v, false, b1.v, (short)0, acc[1][1], false, false);
    acc[1][2] = __builtin_amdgcn_wmma_f32_16x16x32_bf16(false, a1.v, false, b2.v, (short)0, acc[1][2], false, false);
    acc[1][3] = __builtin_amdgcn_wmma_f32_16x16x32_bf16(false, a1.v, false, b3.v, (short)0, acc[1][3], false, false);
  }

  // epilogue: C/D layout => lane holds col n = tile_n + ln, rows r = tile_m + kh*8 + v
#pragma unroll
  for (int i = 0; i < 2; ++i) {
    const int r0 = mBase + i * 16 + kh * 8;
    float4 bs0 = *(const float4*)(bias + r0);
    float4 bs1 = *(const float4*)(bias + r0 + 4);
    float bb[8] = {bs0.x, bs0.y, bs0.z, bs0.w, bs1.x, bs1.y, bs1.z, bs1.w};
    float sc[8], sh[8];
    if (bng != nullptr) {
      float4 g0 = *(const float4*)(bng + r0), g1 = *(const float4*)(bng + r0 + 4);
      float4 e0 = *(const float4*)(bnb + r0), e1 = *(const float4*)(bnb + r0 + 4);
      float4 m0 = *(const float4*)(bnm + r0), m1 = *(const float4*)(bnm + r0 + 4);
      float4 v0 = *(const float4*)(bnv + r0), v1 = *(const float4*)(bnv + r0 + 4);
      float gg[8] = {g0.x, g0.y, g0.z, g0.w, g1.x, g1.y, g1.z, g1.w};
      float ee[8] = {e0.x, e0.y, e0.z, e0.w, e1.x, e1.y, e1.z, e1.w};
      float mm[8] = {m0.x, m0.y, m0.z, m0.w, m1.x, m1.y, m1.z, m1.w};
      float vv[8] = {v0.x, v0.y, v0.z, v0.w, v1.x, v1.y, v1.z, v1.w};
#pragma unroll
      for (int e = 0; e < 8; ++e) {
        float s = gg[e] * rsqrtf(vv[e] + BN_EPS);
        sc[e] = s;
        sh[e] = ee[e] + (bb[e] - mm[e]) * s;   // y = (acc+bias)*s + (be - m*s)
      }
    } else {
#pragma unroll
      for (int e = 0; e < 8; ++e) { sc[e] = 1.f; sh[e] = bb[e]; }
    }
#pragma unroll
    for (int j = 0; j < 4; ++j) {
      const int n = nBase + j * 16 + ln;
      float o[8];
#pragma unroll
      for (int e = 0; e < 8; ++e) {
        o[e] = acc[i][j][e] * sc[e] + sh[e];
        if (bng != nullptr) o[e] = fmaxf(o[e], 0.f);
      }
      if (mode == 0) {
        float* d = outF + (long long)blockIdx.z * N * M + (long long)n * M + r0;
        *(float4*)(d)     = make_float4(o[0], o[1], o[2], o[3]);
        *(float4*)(d + 4) = make_float4(o[4], o[5], o[6], o[7]);
      } else if (mode == 1) {
        Pack8 pk;
#pragma unroll
        for (int e = 0; e < 8; ++e) pk.u[e] = f2bf(o[e]);
        *(uint4*)(outB + (long long)blockIdx.z * N * M + (long long)n * M + r0) = pk.q;
      } else {
        float* d = outF + (long long)blockIdx.z * M * N + n;
#pragma unroll
        for (int e = 0; e < 8; ++e) d[(long long)(r0 + e) * N] = o[e];
      }
    }
  }
}

// ---------------------------------------------------------------------------
// 27-tap dilated local attention. One wave per pixel (8 pixels / 256-thr block).
// keyT/queryT: [B*HW][256] bf16 ; valueT: [B*HW][256] f32 ; ctxT: [B*HW][256] bf16
// ---------------------------------------------------------------------------
__global__ void __launch_bounds__(256)
attn_kernel(const unsigned short* __restrict__ keyT,
            const unsigned short* __restrict__ queryT,
            const float* __restrict__ valueT,
            unsigned short* __restrict__ ctxT) {
  const int wave = threadIdx.x >> 5, lane = threadIdx.x & 31;
  const int gp = blockIdx.x * 8 + wave;   // global pixel 0..16383
  const int b = gp >> 12;
  const int n = gp & 4095;
  const int h = n >> 6, w = n & 63;
  const int c0 = lane * 8;                // 8 contiguous channels per lane

  float q[8];
  {
    Pack8 pq;
    pq.q = *(const uint4*)(queryT + (long long)gp * 256 + c0);
#pragma unroll
    for (int e = 0; e < 8; ++e) q[e] = bf2f(pq.u[e]);
  }

  float sim[27];
#pragma unroll
  for (int dd = 0; dd < 3; ++dd) {
    const int d = 1 << dd;  // dilations 1,2,4
#pragma unroll
    for (int i = 0; i < 3; ++i) {
#pragma unroll
      for (int j = 0; j < 3; ++j) {
        const int t = dd * 9 + i * 3 + j;
        const int hh = h + (i - 1) * d, ww = w + (j - 1) * d;
        float dot = 0.f;
        if ((unsigned)hh < 64u && (unsigned)ww < 64u) {  // zero-padded taps -> sim 0
          const int nb = (b << 12) + (hh << 6) + ww;
          Pack8 pk;
          pk.q = *(const uint4*)(keyT + (long long)nb * 256 + c0);
#pragma unroll
          for (int e = 0; e < 8; ++e) dot += q[e] * bf2f(pk.u[e]);
        }
#pragma unroll
        for (int off = 16; off > 0; off >>= 1)   // wave32 reduction
          dot += __shfl_xor(dot, off, 32);
        sim[t] = dot;
      }
    }
  }

  float mx = sim[0];
#pragma unroll
  for (int t = 1; t < 27; ++t) mx = fmaxf(mx, sim[t]);
  float ssum = 0.f;
#pragma unroll
  for (int t = 0; t < 27; ++t) { sim[t] = __expf(sim[t] - mx); ssum += sim[t]; }
  const float inv = 1.f / ssum;

  float ctx[8] = {0.f, 0.f, 0.f, 0.f, 0.f, 0.f, 0.f, 0.f};
#pragma unroll
  for (int dd = 0; dd < 3; ++dd) {
    const int d = 1 << dd;
#pragma unroll
    for (int i = 0; i < 3; ++i) {
#pragma unroll
      for (int j = 0; j < 3; ++j) {
        const int t = dd * 9 + i * 3 + j;
        const int hh = h + (i - 1) * d, ww = w + (j - 1) * d;
        if ((unsigned)hh < 64u && (unsigned)ww < 64u) {
          const int nb = (b << 12) + (hh << 6) + ww;
          const float wt = sim[t] * inv;
          float4 v0 = *(const float4*)(valueT + (long long)nb * 256 + c0);
          float4 v1 = *(const float4*)(valueT + (long long)nb * 256 + c0 + 4);
          ctx[0] += wt * v0.x; ctx[1] += wt * v0.y; ctx[2] += wt * v0.z; ctx[3] += wt * v0.w;
          ctx[4] += wt * v1.x; ctx[5] += wt * v1.y; ctx[6] += wt * v1.z; ctx[7] += wt * v1.w;
        }
      }
    }
  }

  Pack8 pc;
#pragma unroll
  for (int e = 0; e < 8; ++e) pc.u[e] = f2bf(ctx[e]);
  *(uint4*)(ctxT + (long long)gp * 256 + c0) = pc.q;
}

// ---------------------------------------------------------------------------
extern "C" void kernel_launch(void* const* d_in, const int* in_sizes, int n_in,
                              void* d_out, int out_size, void* d_ws, size_t ws_size,
                              hipStream_t stream) {
  (void)in_sizes; (void)n_in; (void)out_size; (void)ws_size;

  const float* x     = (const float*)d_in[0];
  const float* k1_w  = (const float*)d_in[1];
  const float* k1_b  = (const float*)d_in[2];
  const float* k1_g  = (const float*)d_in[3];
  const float* k1_be = (const float*)d_in[4];
  const float* k1_m  = (const float*)d_in[5];
  const float* k1_v  = (const float*)d_in[6];
  const float* k2_w  = (const float*)d_in[7];
  const float* k2_b  = (const float*)d_in[8];
  const float* k2_g  = (const float*)d_in[9];
  const float* k2_be = (const float*)d_in[10];
  const float* k2_m  = (const float*)d_in[11];
  const float* k2_v  = (const float*)d_in[12];
  const float* q1_w  = (const float*)d_in[13];
  const float* q1_b  = (const float*)d_in[14];
  const float* q1_g  = (const float*)d_in[15];
  const float* q1_be = (const float*)d_in[16];
  const float* q1_m  = (const float*)d_in[17];
  const float* q1_v  = (const float*)d_in[18];
  const float* q2_w  = (const float*)d_in[19];
  const float* q2_b  = (const float*)d_in[20];
  const float* q2_g  = (const float*)d_in[21];
  const float* q2_be = (const float*)d_in[22];
  const float* q2_m  = (const float*)d_in[23];
  const float* q2_v  = (const float*)d_in[24];
  const float* v_w   = (const float*)d_in[25];
  const float* v_b   = (const float*)d_in[26];
  const float* w_w   = (const float*)d_in[27];
  const float* w_b   = (const float*)d_in[28];

  const int B = 4, Cin = 512, Ck = 256, Cv = 256, Cout = 512, HW = 4096;

  char* ws = (char*)d_ws;
  size_t off = 0;
  auto alloc = [&](size_t bytes) -> void* {
    void* p = ws + off;
    off = (off + bytes + 255) & ~(size_t)255;
    return p;
  };
  unsigned short* xbT    = (unsigned short*)alloc((size_t)B * HW * Cin * 2);  // x^T bf16
  float*          valT   = (float*)         alloc((size_t)B * HW * Cv  * 4); // value^T f32
  unsigned short* khT    = (unsigned short*)alloc((size_t)B * HW * Ck * 2);
  unsigned short* keyT   = (unsigned short*)alloc((size_t)B * HW * Ck * 2);
  unsigned short* qhT    = (unsigned short*)alloc((size_t)B * HW * Ck * 2);
  unsigned short* queryT = (unsigned short*)alloc((size_t)B * HW * Ck * 2);
  unsigned short* ctxT   = (unsigned short*)alloc((size_t)B * HW * Cv * 2);
  unsigned short* wv  = (unsigned short*)alloc((size_t)Cv * Cin * 2);
  unsigned short* wk1 = (unsigned short*)alloc((size_t)Ck * Cin * 2);
  unsigned short* wk2 = (unsigned short*)alloc((size_t)Ck * Ck  * 2);
  unsigned short* wq1 = (unsigned short*)alloc((size_t)Ck * Cin * 2);
  unsigned short* wq2 = (unsigned short*)alloc((size_t)Ck * Ck  * 2);
  unsigned short* ww  = (unsigned short*)alloc((size_t)Cout * Cv * 2);

  // weights -> bf16
  cast_bf16_kernel<<<(Cv * Cin + 255) / 256, 256, 0, stream>>>(v_w, wv, Cv * Cin);
  cast_bf16_kernel<<<(Ck * Cin + 255) / 256, 256, 0, stream>>>(k1_w, wk1, Ck * Cin);
  cast_bf16_kernel<<<(Ck * Ck + 255) / 256, 256, 0, stream>>>(k2_w, wk2, Ck * Ck);
  cast_bf16_kernel<<<(Ck * Cin + 255) / 256, 256, 0, stream>>>(q1_w, wq1, Ck * Cin);
  cast_bf16_kernel<<<(Ck * Ck + 255) / 256, 256, 0, stream>>>(q2_w, wq2, Ck * Ck);
  cast_bf16_kernel<<<(Cout * Cv + 255) / 256, 256, 0, stream>>>(w_w, ww, Cout * Cv);

  // x -> pixel-major bf16 (LDS-tiled transpose)
  transpose_cast_kernel<<<dim3(HW / 32, Cin / 32, B), 256, 0, stream>>>(x, xbT, Cin, HW);

  // value = conv1x1(x, v_w) + v_b           -> f32 [b][n][Cv]
  gemm_wmma_bf16<<<dim3(HW / 256, Cv / 64, B), 256, 0, stream>>>(
      wv, xbT, valT, nullptr, v_b, nullptr, nullptr, nullptr, nullptr, Cv, Cin, HW, 0);
  // key path: conv+BN+ReLU twice           -> bf16 [b][n][Ck]
  gemm_wmma_bf16<<<dim3(HW / 256, Ck / 64, B), 256, 0, stream>>>(
      wk1, xbT, nullptr, khT, k1_b, k1_g, k1_be, k1_m, k1_v, Ck, Cin, HW, 1);
  gemm_wmma_bf16<<<dim3(HW / 256, Ck / 64, B), 256, 0, stream>>>(
      wk2, khT, nullptr, keyT, k2_b, k2_g, k2_be, k2_m, k2_v, Ck, Ck, HW, 1);
  // query path
  gemm_wmma_bf16<<<dim3(HW / 256, Ck / 64, B), 256, 0, stream>>>(
      wq1, xbT, nullptr, qhT, q1_b, q1_g, q1_be, q1_m, q1_v, Ck, Cin, HW, 1);
  gemm_wmma_bf16<<<dim3(HW / 256, Ck / 64, B), 256, 0, stream>>>(
      wq2, qhT, nullptr, queryT, q2_b, q2_g, q2_be, q2_m, q2_v, Ck, Ck, HW, 1);

  // 27-tap dilated attention -> ctxT bf16 [b][n][Cv]
  attn_kernel<<<(B * HW) / 8, 256, 0, stream>>>(keyT, queryT, valT, ctxT);

  // out = conv1x1(context, w_w) + w_b -> f32 NCHW into d_out
  gemm_wmma_bf16<<<dim3(HW / 256, Cout / 64, B), 256, 0, stream>>>(
      ww, ctxT, (float*)d_out, nullptr, w_b, nullptr, nullptr, nullptr, nullptr, Cout, Cv, HW, 2);
}
